// Seq2SeqAttnModel_45896020525970
// MI455X (gfx1250) — matrix operand order repaired and checked by reading, not compiled, not run on hardware
//
#include <hip/hip_runtime.h>

// ---------------------------------------------------------------- types
typedef unsigned short u16;
typedef __attribute__((ext_vector_type(16))) __bf16 bf16x16;
typedef __attribute__((ext_vector_type(8)))  float  f32x8;
typedef __attribute__((ext_vector_type(8)))  u16    u16x8;

union FragU { bf16x16 v; u16x8 h[2]; };

static constexpr int BB  = 64;     // batch
static constexpr int TS  = 50;
static constexpr int TD  = 50;
static constexpr int DSZ = 512;
static constexpr int HSZ = 1024;
static constexpr int G3  = 3 * HSZ;   // 3072
static constexpr int VV  = 32001;

// ---------------------------------------------------------------- helpers
__device__ __forceinline__ float bf2f(u16 h) {
  unsigned int u = ((unsigned int)h) << 16;
  float f; __builtin_memcpy(&f, &u, 4); return f;
}
__device__ __forceinline__ u16 f2bf(float f) {
  unsigned int u; __builtin_memcpy(&u, &f, 4);
  u += 0x7fffu + ((u >> 16) & 1u);
  return (u16)(u >> 16);
}
__device__ __forceinline__ f32x8 zero8() {
  f32x8 z = {0.f,0.f,0.f,0.f,0.f,0.f,0.f,0.f};
  return z;
}
__device__ __forceinline__ float warp_sum(float v) {
  for (int o = 16; o > 0; o >>= 1) v += __shfl_down(v, o);
  return v;
}
__device__ __forceinline__ float warp_max(float v) {
  for (int o = 16; o > 0; o >>= 1) v = fmaxf(v, __shfl_down(v, o));
  return v;
}
__device__ __forceinline__ float sigmoidf(float x) { return 1.f / (1.f + __expf(-x)); }

// Multi-N-tile WMMA K-loop with explicit 1-stage software pipeline:
// fragments for step k+1 are loaded before the WMMAs of step k are issued,
// so each WMMA waits only on loads issued a full iteration earlier.
// A frag (16-bit 16x32): lane kh=lane>>4 -> K chunks [kh*8..+7] and [16+kh*8..+7]
// B frag (16-bit 32x16): lane kh -> K chunks [kh*8..+7] and [16+kh*8..+7]
template <int NT>
__device__ __forceinline__ void wmma_kloop_nt(const u16* __restrict__ ap,
                                              const u16* const (&wp)[NT],
                                              int K, int kh, f32x8 (&acc)[NT]) {
  FragU a; FragU b[NT];
  a.h[0] = *(const u16x8*)(ap + kh * 8);
  a.h[1] = *(const u16x8*)(ap + 16 + kh * 8);
#pragma unroll
  for (int i = 0; i < NT; ++i) {
    b[i].h[0] = *(const u16x8*)(wp[i] + kh * 8);
    b[i].h[1] = *(const u16x8*)(wp[i] + 16 + kh * 8);
  }
  for (int k0 = 32; k0 < K; k0 += 32) {
    __builtin_prefetch(ap + k0 + 256, 0, 1);
    FragU an; FragU bn[NT];
    an.h[0] = *(const u16x8*)(ap + k0 + kh * 8);
    an.h[1] = *(const u16x8*)(ap + k0 + 16 + kh * 8);
#pragma unroll
    for (int i = 0; i < NT; ++i) {
      __builtin_prefetch(wp[i] + k0 + 256, 0, 1);
      bn[i].h[0] = *(const u16x8*)(wp[i] + k0 + kh * 8);
      bn[i].h[1] = *(const u16x8*)(wp[i] + k0 + 16 + kh * 8);
    }
#pragma unroll
    for (int i = 0; i < NT; ++i)
      acc[i] = __builtin_amdgcn_wmma_f32_16x16x32_bf16(
          false, a.v, false, b[i].v, (short)0, acc[i], false, false);
    a = an;
#pragma unroll
    for (int i = 0; i < NT; ++i) b[i] = bn[i];
  }
#pragma unroll
  for (int i = 0; i < NT; ++i)
    acc[i] = __builtin_amdgcn_wmma_f32_16x16x32_bf16(
        false, a.v, false, b[i].v, (short)0, acc[i], false, false);
}

// ---------------------------------------------------------------- conversions
__global__ void k_f32_to_bf16(const float* __restrict__ s, u16* __restrict__ d, long n) {
  long i = (long)blockIdx.x * blockDim.x + threadIdx.x;
  if (i < n) d[i] = f2bf(s[i]);
}

// out[t][b][d] = bf16(table[idx[b*T+t]][d])
__global__ void k_embed_gather(const float* __restrict__ table, const int* __restrict__ idx,
                               u16* __restrict__ out, int T, int D) {
  long i = (long)blockIdx.x * blockDim.x + threadIdx.x;
  long total = (long)T * BB * D;
  if (i >= total) return;
  int d = (int)(i % D);
  long r = i / D;
  int b = (int)(r % BB);
  int t = (int)(r / BB);
  int row = idx[b * T + t];
  out[i] = f2bf(table[(size_t)row * D + d]);
}

// ---------------------------------------------------------------- GRU gates GEMM
// gi = X @ Wih^T (+bih) and gh = Hb @ Whh^T (+bhh), both [64][3072], bf16 in, f32 out.
// grid.x = 192 (96 gi super-tiles + 96 gh super-tiles of 32 cols), block = 128
// (4 waves; wave w -> mtile w of M=64; each wave: 2 N-tiles sharing one A frag)
__global__ void k_gru_gates(const u16* __restrict__ X, int Kx,
                            const u16* __restrict__ Wih, const float* __restrict__ bih,
                            const u16* __restrict__ Hb,
                            const u16* __restrict__ Whh, const float* __restrict__ bhh,
                            float* __restrict__ gi, float* __restrict__ gh) {
  int wave = threadIdx.x >> 5;
  int lane = threadIdx.x & 31;
  int nt = blockIdx.x;
  const u16* A; const u16* W; const float* bias; float* C; int K; int n0;
  if (nt < G3 / 32) { A = X;  W = Wih; bias = bih; C = gi; K = Kx;  n0 = nt * 32; }
  else              { A = Hb; W = Whh; bias = bhh; C = gh; K = HSZ; n0 = (nt - G3 / 32) * 32; }
  int mr = lane & 15, kh = lane >> 4;
  int m = wave * 16 + mr;
  const u16* ap = A + (size_t)m * K;
  const u16* wp[2] = { W + (size_t)(n0 + mr) * K, W + (size_t)(n0 + 16 + mr) * K };
  f32x8 acc[2] = { zero8(), zero8() };
  wmma_kloop_nt<2>(ap, wp, K, kh, acc);
#pragma unroll
  for (int i = 0; i < 2; ++i) {
    int n = n0 + i * 16 + mr;
    float bv = bias[n];
#pragma unroll
    for (int r = 0; r < 8; ++r) {
      int row = wave * 16 + 8 * kh + r;
      C[(size_t)row * G3 + n] = acc[i][r] + bv;
    }
  }
}

// ---------------------------------------------------------------- GRU pointwise
__global__ void k_gru_pointwise(const float* __restrict__ gi, const float* __restrict__ gh,
                                const float* __restrict__ hprev,
                                float* __restrict__ hnew, u16* __restrict__ hnew_bf,
                                u16* __restrict__ ybf) {
  int i = blockIdx.x * blockDim.x + threadIdx.x;
  if (i >= BB * HSZ) return;
  int b = i >> 10, j = i & (HSZ - 1);
  size_t g = (size_t)b * G3 + j;
  float r = sigmoidf(gi[g]            + gh[g]);
  float z = sigmoidf(gi[g + HSZ]      + gh[g + HSZ]);
  float nn = tanhf(  gi[g + 2 * HSZ] + r * gh[g + 2 * HSZ]);
  float h = (1.f - z) * nn + z * hprev[i];
  hnew[i] = h;
  hnew_bf[i] = f2bf(h);
  if (ybf) ybf[i] = f2bf(h);
}

// ---------------------------------------------------------------- decoder input concat
__global__ void k_concat(const u16* __restrict__ emb_t, const u16* __restrict__ outbf,
                         u16* __restrict__ x0) {
  int i = blockIdx.x * blockDim.x + threadIdx.x;
  if (i >= BB * (DSZ + HSZ)) return;
  int b = i / (DSZ + HSZ), j = i % (DSZ + HSZ);
  x0[i] = (j < DSZ) ? emb_t[b * DSZ + j] : outbf[b * HSZ + (j - DSZ)];
}

// ---------------------------------------------------------------- generic WMMA GEMM
// C[M][N] = A[M][K] @ W[N][K]^T (+bias), bf16 in, f32 out.
// grid=(ceil(N/32), ceil(M/64)), block=128; each wave: 2 N-tiles, shared A frag.
__global__ void k_wmma_gemm(const u16* __restrict__ A, const u16* __restrict__ W,
                            const float* __restrict__ bias, float* __restrict__ C,
                            int M, int N, int K) {
  int wave = threadIdx.x >> 5;
  int lane = threadIdx.x & 31;
  int mtile = blockIdx.y * 4 + wave;
  if (mtile * 16 >= M) return;
  int n0 = blockIdx.x * 32;
  int mr = lane & 15, kh = lane >> 4;
  int m = mtile * 16 + mr;
  const u16* ap = A + (size_t)m * K;
  int nn[2]; const u16* wp[2];
#pragma unroll
  for (int i = 0; i < 2; ++i) {
    int n = n0 + i * 16 + mr;
    nn[i] = n;
    wp[i] = W + (size_t)(n < N ? n : 0) * K;
  }
  f32x8 acc[2] = { zero8(), zero8() };
  wmma_kloop_nt<2>(ap, wp, K, kh, acc);
#pragma unroll
  for (int i = 0; i < 2; ++i) {
    if (nn[i] < N) {
      float bv = bias ? bias[nn[i]] : 0.f;
#pragma unroll
      for (int r = 0; r < 8; ++r) {
        int row = mtile * 16 + 8 * kh + r;
        C[(size_t)row * N + nn[i]] = acc[i][r] + bv;
      }
    }
  }
}

// ---------------------------------------------------------------- logits GEMM -> d_out
// A = outs bf16 [TD*BB][HSZ] (row = t*64+b), W = preds bf16 [VV][HSZ], out[b][t][v] f32
// grid = (ceil(VV/64)=501, (TD*BB)/64 = 50), block = 128; each wave: 4 N-tiles.
__global__ void k_wmma_gemm_logits(const u16* __restrict__ A, const u16* __restrict__ W,
                                   const float* __restrict__ bias, float* __restrict__ Out) {
  const int K = HSZ;
  int wave = threadIdx.x >> 5;
  int lane = threadIdx.x & 31;
  int mtile = blockIdx.y * 4 + wave;
  int n0 = blockIdx.x * 64;
  int mr = lane & 15, kh = lane >> 4;
  int m = mtile * 16 + mr;
  const u16* ap = A + (size_t)m * K;
  int nn[4]; const u16* wp[4];
#pragma unroll
  for (int i = 0; i < 4; ++i) {
    int n = n0 + i * 16 + mr;
    nn[i] = n;
    wp[i] = W + (size_t)(n < VV ? n : 0) * K;
  }
  f32x8 acc[4] = { zero8(), zero8(), zero8(), zero8() };
  wmma_kloop_nt<4>(ap, wp, K, kh, acc);
#pragma unroll
  for (int i = 0; i < 4; ++i) {
    if (nn[i] < VV) {
      float bv = bias[nn[i]];
#pragma unroll
      for (int r = 0; r < 8; ++r) {
        int row = mtile * 16 + 8 * kh + r;   // = t*64 + b
        int t = row >> 6, b = row & 63;
        Out[((size_t)b * TD + t) * VV + nn[i]] = acc[i][r] + bv;
      }
    }
  }
}

// ---------------------------------------------------------------- attention
// one block per batch row; writes concat2[b] = [comb bf16 (1024) | x bf16 (1024)]
__global__ void k_attention(const float* __restrict__ q, const u16* __restrict__ ctx,
                            const u16* __restrict__ xbf, u16* __restrict__ concat2) {
  __shared__ float sc[TS];
  __shared__ float sa[TS];
  int b = blockIdx.x;
  int tid = threadIdx.x, lane = tid & 31, wave = tid >> 5;
  const float* qb = q + (size_t)b * HSZ;
  for (int t = wave; t < TS; t += 8) {
    const u16* c = ctx + ((size_t)t * BB + b) * HSZ;
    float s = 0.f;
    for (int h = lane; h < HSZ; h += 32) s += bf2f(c[h]) * qb[h];
    s = warp_sum(s);
    if (lane == 0) sc[t] = s;
  }
  __syncthreads();
  if (tid == 0) {
    float mx = sc[0];
    for (int t = 1; t < TS; ++t) mx = fmaxf(mx, sc[t]);
    float sum = 0.f;
    for (int t = 0; t < TS; ++t) { float e = __expf(sc[t] - mx); sa[t] = e; sum += e; }
    float inv = 1.f / sum;
    for (int t = 0; t < TS; ++t) sa[t] *= inv;
  }
  __syncthreads();
  for (int h = tid; h < HSZ; h += 256) {
    float acc = 0.f;
    for (int t = 0; t < TS; ++t)
      acc += sa[t] * bf2f(ctx[((size_t)t * BB + b) * HSZ + h]);
    concat2[(size_t)b * 2 * HSZ + h] = f2bf(acc);
    concat2[(size_t)b * 2 * HSZ + HSZ + h] = xbf[(size_t)b * HSZ + h];
  }
}

// ---------------------------------------------------------------- tanh epilogue
__global__ void k_tanh_out(const float* __restrict__ g, u16* __restrict__ outbf,
                           u16* __restrict__ outs_t) {
  int i = blockIdx.x * blockDim.x + threadIdx.x;
  if (i >= BB * HSZ) return;
  float v = tanhf(g[i]);
  u16 h = f2bf(v);
  outbf[i] = h;
  outs_t[i] = h;
}

// ---------------------------------------------------------------- log_softmax (in place)
__global__ void k_log_softmax(float* __restrict__ out) {
  float* row = out + (size_t)blockIdx.x * VV;
  __shared__ float red[8];
  int tid = threadIdx.x, lane = tid & 31, wave = tid >> 5;
  float mx = -3.0e38f;
  for (int v = tid; v < VV; v += 256) mx = fmaxf(mx, row[v]);
  mx = warp_max(mx);
  if (lane == 0) red[wave] = mx;
  __syncthreads();
  mx = red[0];
  for (int w = 1; w < 8; ++w) mx = fmaxf(mx, red[w]);
  __syncthreads();
  float s = 0.f;
  for (int v = tid; v < VV; v += 256) s += __expf(row[v] - mx);
  s = warp_sum(s);
  if (lane == 0) red[wave] = s;
  __syncthreads();
  s = 0.f;
  for (int w = 0; w < 8; ++w) s += red[w];
  float lse = mx + __logf(s);
  for (int v = tid; v < VV; v += 256) row[v] -= lse;
}

// ---------------------------------------------------------------- host orchestration
extern "C" void kernel_launch(void* const* d_in, const int* in_sizes, int n_in,
                              void* d_out, int out_size, void* d_ws, size_t ws_size,
                              hipStream_t stream) {
  (void)in_sizes; (void)n_in; (void)out_size; (void)ws_size;
  const int*   src       = (const int*)d_in[0];
  const int*   dst       = (const int*)d_in[1];
  const float* embed_in  = (const float*)d_in[2];
  const float* embed_out = (const float*)d_in[3];
  const float* e_wih[2]  = {(const float*)d_in[4],  (const float*)d_in[8]};
  const float* e_whh[2]  = {(const float*)d_in[5],  (const float*)d_in[9]};
  const float* e_bih[2]  = {(const float*)d_in[6],  (const float*)d_in[10]};
  const float* e_bhh[2]  = {(const float*)d_in[7],  (const float*)d_in[11]};
  const float* d_wih[2]  = {(const float*)d_in[12], (const float*)d_in[16]};
  const float* d_whh[2]  = {(const float*)d_in[13], (const float*)d_in[17]};
  const float* d_bih[2]  = {(const float*)d_in[14], (const float*)d_in[18]};
  const float* d_bhh[2]  = {(const float*)d_in[15], (const float*)d_in[19]};
  const float* w_o2a     = (const float*)d_in[20];
  const float* w_attnout = (const float*)d_in[21];
  const float* w_preds   = (const float*)d_in[22];
  const float* b_preds   = (const float*)d_in[23];
  float* out = (float*)d_out;

  // workspace allocator
  char* ws = (char*)d_ws;
  size_t off = 0;
  auto alloc = [&](size_t bytes) -> void* {
    void* p = ws + off;
    off += (bytes + 255) & ~(size_t)255;
    return p;
  };
  // bf16 weights
  u16* ewih0 = (u16*)alloc((size_t)G3 * DSZ * 2);
  u16* ewhh0 = (u16*)alloc((size_t)G3 * HSZ * 2);
  u16* ewih1 = (u16*)alloc((size_t)G3 * HSZ * 2);
  u16* ewhh1 = (u16*)alloc((size_t)G3 * HSZ * 2);
  u16* dwih0 = (u16*)alloc((size_t)G3 * (DSZ + HSZ) * 2);
  u16* dwhh0 = (u16*)alloc((size_t)G3 * HSZ * 2);
  u16* dwih1 = (u16*)alloc((size_t)G3 * HSZ * 2);
  u16* dwhh1 = (u16*)alloc((size_t)G3 * HSZ * 2);
  u16* o2ab  = (u16*)alloc((size_t)HSZ * HSZ * 2);
  u16* attnb = (u16*)alloc((size_t)HSZ * 2 * HSZ * 2);
  u16* predb = (u16*)alloc((size_t)VV * HSZ * 2);
  // bf16 activations
  u16* xs    = (u16*)alloc((size_t)TS * BB * DSZ * 2);
  u16* ys0   = (u16*)alloc((size_t)TS * BB * HSZ * 2);
  u16* ctx   = (u16*)alloc((size_t)TS * BB * HSZ * 2);
  u16* embd  = (u16*)alloc((size_t)TD * BB * DSZ * 2);
  u16* outs  = (u16*)alloc((size_t)TD * BB * HSZ * 2);
  u16* h_enc_b  = (u16*)alloc((size_t)BB * HSZ * 2);
  u16* h_dec0_b = (u16*)alloc((size_t)BB * HSZ * 2);
  u16* h_dec1_b = (u16*)alloc((size_t)BB * HSZ * 2);
  u16* out_bf   = (u16*)alloc((size_t)BB * HSZ * 2);
  u16* x0       = (u16*)alloc((size_t)BB * (DSZ + HSZ) * 2);
  u16* concat2  = (u16*)alloc((size_t)BB * 2 * HSZ * 2);
  // f32 scratch
  float* gi = (float*)alloc((size_t)BB * G3 * 4);
  float* gh = (float*)alloc((size_t)BB * G3 * 4);
  float* h_enc_f  = (float*)alloc((size_t)BB * HSZ * 4);
  float* h_dec0_f = (float*)alloc((size_t)BB * HSZ * 4);
  float* h_dec1_f = (float*)alloc((size_t)BB * HSZ * 4);
  float* qbuf     = (float*)alloc((size_t)BB * HSZ * 4);
  float* gemmout  = (float*)alloc((size_t)BB * HSZ * 4);

  auto conv = [&](const float* s, u16* d, long n) {
    int blocks = (int)((n + 255) / 256);
    k_f32_to_bf16<<<blocks, 256, 0, stream>>>(s, d, n);
  };
  // weight conversion (every call; deterministic)
  conv(e_wih[0], ewih0, (long)G3 * DSZ);
  conv(e_whh[0], ewhh0, (long)G3 * HSZ);
  conv(e_wih[1], ewih1, (long)G3 * HSZ);
  conv(e_whh[1], ewhh1, (long)G3 * HSZ);
  conv(d_wih[0], dwih0, (long)G3 * (DSZ + HSZ));
  conv(d_whh[0], dwhh0, (long)G3 * HSZ);
  conv(d_wih[1], dwih1, (long)G3 * HSZ);
  conv(d_whh[1], dwhh1, (long)G3 * HSZ);
  conv(w_o2a, o2ab, (long)HSZ * HSZ);
  conv(w_attnout, attnb, (long)HSZ * 2 * HSZ);
  conv(w_preds, predb, (long)VV * HSZ);

  // embedding gathers (time-major, bf16)
  {
    long n = (long)TS * BB * DSZ;
    k_embed_gather<<<(int)((n + 255) / 256), 256, 0, stream>>>(embed_in, src, xs, TS, DSZ);
    n = (long)TD * BB * DSZ;
    k_embed_gather<<<(int)((n + 255) / 256), 256, 0, stream>>>(embed_out, dst, embd, TD, DSZ);
  }

  const dim3 gatesGrid(2 * G3 / 32);    // 192 (96 gi + 96 gh super-tiles)
  const dim3 gemm64(HSZ / 32, 1);       // N=1024 -> 32 super-tiles, M=64

  // ---- encoder layer 0 ----
  hipMemsetAsync(h_enc_f, 0, (size_t)BB * HSZ * 4, stream);
  hipMemsetAsync(h_enc_b, 0, (size_t)BB * HSZ * 2, stream);
  for (int t = 0; t < TS; ++t) {
    k_gru_gates<<<gatesGrid, 128, 0, stream>>>(
        xs + (size_t)t * BB * DSZ, DSZ, ewih0, e_bih[0], h_enc_b, ewhh0, e_bhh[0], gi, gh);
    k_gru_pointwise<<<(BB * HSZ) / 256, 256, 0, stream>>>(
        gi, gh, h_enc_f, h_enc_f, h_enc_b, ys0 + (size_t)t * BB * HSZ);
  }
  hipMemcpyAsync(h_dec0_f, h_enc_f, (size_t)BB * HSZ * 4, hipMemcpyDeviceToDevice, stream);
  hipMemcpyAsync(h_dec0_b, h_enc_b, (size_t)BB * HSZ * 2, hipMemcpyDeviceToDevice, stream);

  // ---- encoder layer 1 ----
  hipMemsetAsync(h_enc_f, 0, (size_t)BB * HSZ * 4, stream);
  hipMemsetAsync(h_enc_b, 0, (size_t)BB * HSZ * 2, stream);
  for (int t = 0; t < TS; ++t) {
    k_gru_gates<<<gatesGrid, 128, 0, stream>>>(
        ys0 + (size_t)t * BB * HSZ, HSZ, ewih1, e_bih[1], h_enc_b, ewhh1, e_bhh[1], gi, gh);
    k_gru_pointwise<<<(BB * HSZ) / 256, 256, 0, stream>>>(
        gi, gh, h_enc_f, h_enc_f, h_enc_b, ctx + (size_t)t * BB * HSZ);
  }
  hipMemcpyAsync(h_dec1_f, h_enc_f, (size_t)BB * HSZ * 4, hipMemcpyDeviceToDevice, stream);
  hipMemcpyAsync(h_dec1_b, h_enc_b, (size_t)BB * HSZ * 2, hipMemcpyDeviceToDevice, stream);

  // ---- decoder ----
  hipMemsetAsync(out_bf, 0, (size_t)BB * HSZ * 2, stream);
  for (int t = 0; t < TD; ++t) {
    k_concat<<<(BB * (DSZ + HSZ) + 255) / 256, 256, 0, stream>>>(
        embd + (size_t)t * BB * DSZ, out_bf, x0);
    // layer 0: x = [emb|out] (K=1536)
    k_gru_gates<<<gatesGrid, 128, 0, stream>>>(
        x0, DSZ + HSZ, dwih0, d_bih[0], h_dec0_b, dwhh0, d_bhh[0], gi, gh);
    k_gru_pointwise<<<(BB * HSZ) / 256, 256, 0, stream>>>(
        gi, gh, h_dec0_f, h_dec0_f, h_dec0_b, nullptr);
    // layer 1: x = new h0
    k_gru_gates<<<gatesGrid, 128, 0, stream>>>(
        h_dec0_b, HSZ, dwih1, d_bih[1], h_dec1_b, dwhh1, d_bhh[1], gi, gh);
    k_gru_pointwise<<<(BB * HSZ) / 256, 256, 0, stream>>>(
        gi, gh, h_dec1_f, h_dec1_f, h_dec1_b, nullptr);
    // q = x @ w_o2a^T
    k_wmma_gemm<<<gemm64, 128, 0, stream>>>(h_dec1_b, o2ab, nullptr, qbuf, BB, HSZ, HSZ);
    // attention -> concat2 = [comb | x]
    k_attention<<<BB, 256, 0, stream>>>(qbuf, ctx, h_dec1_b, concat2);
    // comb' = tanh(concat2 @ w_attnout^T)
    k_wmma_gemm<<<gemm64, 128, 0, stream>>>(concat2, attnb, nullptr, gemmout, BB, HSZ, 2 * HSZ);
    k_tanh_out<<<(BB * HSZ) / 256, 256, 0, stream>>>(
        gemmout, out_bf, outs + (size_t)t * BB * HSZ);
  }

  // ---- logits + log_softmax ----
  k_wmma_gemm_logits<<<dim3((VV + 63) / 64, (TD * BB) / 64), 128, 0, stream>>>(
      outs, predb, b_preds, out);
  k_log_softmax<<<BB * TD, 256, 0, stream>>>(out);
}
